// GAT_42339787604896
// MI455X (gfx1250) — compile-verified, matrix operand order checked by reading, not run on hardware
//
#include <hip/hip_runtime.h>
#include <hip/hip_bf16.h>
#include <math.h>

typedef __attribute__((ext_vector_type(16))) _Float16 v16h;
typedef __attribute__((ext_vector_type(8)))  _Float16 v8h;
typedef __attribute__((ext_vector_type(8)))  float    v8f;

#define HEADS 4
#define CDIM 32
#define FDIM 128
#define NEG_SLOPE 0.2f
#define GAT_EPS 1e-16f

// ---------------------------------------------------------------------------
// Zero per-layer accumulators: agg [N*128], m_enc [N*4] (uint 0 == -inf in the
// monotone encoding), s_sum [N*4].
// ---------------------------------------------------------------------------
__global__ void layer_init(float* __restrict__ agg, unsigned* __restrict__ m_enc,
                           float* __restrict__ s_sum, int Nn) {
  int t = blockIdx.x * blockDim.x + threadIdx.x;
  if (t < Nn * FDIM) agg[t] = 0.0f;
  if (t < Nn * HEADS) { m_enc[t] = 0u; s_sum[t] = 0.0f; }
}

// One-time f32 -> f16 conversion of the layer-0 input features.
__global__ void conv_x(const float* __restrict__ X, _Float16* __restrict__ Xh, int total) {
  int t = blockIdx.x * blockDim.x + threadIdx.x;
  if (t < total) Xh[t] = (_Float16)X[t];
}

// ---------------------------------------------------------------------------
// Pre-pack W (f32 [128][128]) into WMMA B-fragment order, f16:
//   Bp[((ct*32 + lane)*4 + kstep)*16 + i]  with
//   k = kstep*32 + (lane>=16 ? 8 : 0) + (i<8 ? i : i+8),  col = ct*16 + lane%16
// so each lane's B fragment per k-step is one contiguous 32-byte load.
// ---------------------------------------------------------------------------
__global__ void pack_b(const float* __restrict__ W, _Float16* __restrict__ Bp) {
  int t = blockIdx.x * blockDim.x + threadIdx.x;
  if (t >= 8 * 32 * 4 * 16) return;
  const int i    = t & 15;
  const int ks   = (t >> 4) & 3;
  const int lane = (t >> 6) & 31;
  const int ct   = t >> 11;
  const int ka   = ks * 32 + ((lane >= 16) ? 8 : 0);
  const int k    = ka + ((i < 8) ? i : i + 8);
  const int col  = ct * 16 + (lane & 15);
  Bp[t] = (_Float16)W[(size_t)k * FDIM + col];
}

// ---------------------------------------------------------------------------
// h = Xh @ W  (Xh: [N,128] f16 row-major, Bp: pre-packed f16 fragments,
// h: [N,128] f32). One wave per 16x16 tile, K=128 via 4 chained
// v_wmma_f32_16x16x32_f16. A layout (ISA 7.12.2, 16-bit A 16x32): lanes 0-15
// hold row M=lane, K = k0+{0..7,16..23}; lanes 16-31 hold row M=lane-16,
// K = k0+{8..15,24..31}.  C/D: VGPR r -> M=r / M=r+8, N=lane%16.
// Epilogue: uniform full-tile fast path (no per-row exec masking).
// ---------------------------------------------------------------------------
__global__ void gemm_wmma(const _Float16* __restrict__ Xh,
                          const _Float16* __restrict__ Bp,
                          float* __restrict__ Hout, int Nn) {
  const int wave = blockIdx.x * (blockDim.x >> 5) + (threadIdx.x >> 5);
  const int lane = threadIdx.x & 31;
  const int rowTiles = (Nn + 15) >> 4;
  if (wave >= rowTiles * (FDIM / 16)) return;

  const int rt = wave >> 3;
  const int ct = wave & 7;
  const int row0 = rt << 4;
  const int col0 = ct << 4;
  const int lm = lane & 15;
  const bool hi = lane >= 16;

  int arow = row0 + lm;
  if (arow >= Nn) arow = Nn - 1;            // clamp; stores are guarded in tail

  const _Float16* arowp = Xh + (size_t)arow * FDIM + (hi ? 8 : 0);
  const _Float16* bfrag = Bp + ((size_t)(ct * 32 + lane) * 4) * 16;

  v8f c = {};
  #pragma unroll
  for (int ks = 0; ks < 4; ++ks) {
    const _Float16* ap = arowp + ks * 32;
    const v8h alo = *(const v8h*)(ap);       // K = ka .. ka+7
    const v8h ahi = *(const v8h*)(ap + 16);  // K = ka+16 .. ka+23
    v16h a;
    #pragma unroll
    for (int i = 0; i < 8; ++i) { a[i] = alo[i]; a[8 + i] = ahi[i]; }
    const v16h b = *(const v16h*)(bfrag + ks * 16);
    c = __builtin_amdgcn_wmma_f32_16x16x32_f16(
        /*neg_a=*/false, a, /*neg_b=*/false, b,
        /*c_mod=*/(short)0, c, /*reuse_a=*/false, /*reuse_b=*/false);
  }

  const int m0 = hi ? 8 : 0;
  float* outp = Hout + (size_t)(row0 + m0) * FDIM + col0 + lm;
  if (row0 + 16 <= Nn) {
    // Full tile: 8 unguarded, lane-coalesced stores (uniform branch).
    #pragma unroll
    for (int r = 0; r < 8; ++r) outp[(size_t)r * FDIM] = c[r];
  } else {
    #pragma unroll
    for (int r = 0; r < 8; ++r) {
      if (row0 + m0 + r < Nn) outp[(size_t)r * FDIM] = c[r];
    }
  }
}

// ---------------------------------------------------------------------------
// al_src/al_dst = einsum('nhc,hc->nh', h, a) ; one thread per (node, head).
// ---------------------------------------------------------------------------
__global__ void attn_logits(const float* __restrict__ Hh,
                            const float* __restrict__ Asrc,
                            const float* __restrict__ Adst,
                            float* __restrict__ als, float* __restrict__ ald,
                            int Nn) {
  int t = blockIdx.x * blockDim.x + threadIdx.x;
  if (t >= Nn * HEADS) return;
  const int n = t >> 2, hh = t & 3;
  const float* hp = Hh + (size_t)n * FDIM + hh * CDIM;
  const float* as = Asrc + hh * CDIM;
  const float* ad = Adst + hh * CDIM;
  float ss = 0.0f, dd = 0.0f;
  #pragma unroll
  for (int cc = 0; cc < CDIM; ++cc) {
    const float v = hp[cc];
    ss = fmaf(v, as[cc], ss);
    dd = fmaf(v, ad[cc], dd);
  }
  als[t] = ss;
  ald[t] = dd;
}

__device__ __forceinline__ float leaky(float v) {
  return v > 0.0f ? v : v * NEG_SLOPE;
}

// Monotone float->uint encoding: atomicMax(u32) == float max; enc(x) > 0 for
// every real x, so the init value 0 acts as -inf.
__device__ __forceinline__ unsigned enc_f32(float f) {
  unsigned u = __float_as_uint(f);
  return ((int)u >= 0) ? (u | 0x80000000u) : ~u;
}

// ---------------------------------------------------------------------------
// Segment-max of edge logits into m_enc[dst*H+h]. One thread per edge;
// per-node head vectors fetched as float4.
// ---------------------------------------------------------------------------
__global__ void edge_max(const int* __restrict__ src, const int* __restrict__ dst,
                         const float4* __restrict__ als4, const float4* __restrict__ ald4,
                         unsigned* __restrict__ m_enc, int Etot) {
  int e = blockIdx.x * blockDim.x + threadIdx.x;
  if (e >= Etot) return;
  const int si = src[e], di = dst[e];
  const float4 s4 = als4[si];
  const float4 d4 = ald4[di];
  atomicMax(&m_enc[di * HEADS + 0], enc_f32(leaky(s4.x + d4.x)));
  atomicMax(&m_enc[di * HEADS + 1], enc_f32(leaky(s4.y + d4.y)));
  atomicMax(&m_enc[di * HEADS + 2], enc_f32(leaky(s4.z + d4.z)));
  atomicMax(&m_enc[di * HEADS + 3], enc_f32(leaky(s4.w + d4.w)));
}

// In-place decode m_enc (uint) -> m (float); every node has a self loop, so
// every slot was written at least once.
__global__ void decode_max(unsigned* __restrict__ m_enc, int Nn) {
  int t = blockIdx.x * blockDim.x + threadIdx.x;
  if (t >= Nn * HEADS) return;
  const unsigned u = m_enc[t];
  const float f = (u & 0x80000000u) ? __uint_as_float(u ^ 0x80000000u)
                                    : __uint_as_float(~u);
  ((float*)m_enc)[t] = f;
}

// ---------------------------------------------------------------------------
// expe = exp(e - m[dst]) per (edge, head); segment-sum into s_sum.
// ---------------------------------------------------------------------------
__global__ void edge_expsum(const int* __restrict__ src, const int* __restrict__ dst,
                            const float4* __restrict__ als4, const float4* __restrict__ ald4,
                            const float4* __restrict__ m4, float* __restrict__ s_sum,
                            float4* __restrict__ expe4, int Etot) {
  int e = blockIdx.x * blockDim.x + threadIdx.x;
  if (e >= Etot) return;
  const int si = src[e], di = dst[e];
  const float4 s4 = als4[si];
  const float4 d4 = ald4[di];
  const float4 mm = m4[di];
  float4 ex;
  ex.x = __expf(leaky(s4.x + d4.x) - mm.x);
  ex.y = __expf(leaky(s4.y + d4.y) - mm.y);
  ex.z = __expf(leaky(s4.z + d4.z) - mm.z);
  ex.w = __expf(leaky(s4.w + d4.w) - mm.w);
  expe4[e] = ex;
  atomicAdd(&s_sum[di * HEADS + 0], ex.x);
  atomicAdd(&s_sum[di * HEADS + 1], ex.y);
  atomicAdd(&s_sum[di * HEADS + 2], ex.z);
  atomicAdd(&s_sum[di * HEADS + 3], ex.w);
}

// ---------------------------------------------------------------------------
// agg[dst] += h[src] * alpha ; thread per (edge, channel), 4 heads unrolled.
// Working set (h + agg + stats ~ 65 MB) fits in the 192 MB L2, so the random
// gathers and f32 atomics stay on-chip.
// ---------------------------------------------------------------------------
__global__ void edge_aggregate(const int* __restrict__ src, const int* __restrict__ dst,
                               const float* __restrict__ Hh,
                               const float4* __restrict__ s_sum4,
                               const float4* __restrict__ expe4,
                               float* __restrict__ agg, int Etot) {
  int t = blockIdx.x * blockDim.x + threadIdx.x;
  if (t >= Etot * CDIM) return;
  const int e = t >> 5;
  const int cc = t & 31;
  const int si = src[e], di = dst[e];
  const float4 ex = expe4[e];
  const float4 sm = s_sum4[di];
  const float a0 = ex.x / (sm.x + GAT_EPS);
  const float a1 = ex.y / (sm.y + GAT_EPS);
  const float a2 = ex.z / (sm.z + GAT_EPS);
  const float a3 = ex.w / (sm.w + GAT_EPS);
  const float* hp = Hh + (size_t)si * FDIM + cc;
  float* ap = agg + (size_t)di * FDIM + cc;
  atomicAdd(ap + 0 * CDIM, hp[0 * CDIM] * a0);
  atomicAdd(ap + 1 * CDIM, hp[1 * CDIM] * a1);
  atomicAdd(ap + 2 * CDIM, hp[2 * CDIM] * a2);
  atomicAdd(ap + 3 * CDIM, hp[3 * CDIM] * a3);
}

// ---------------------------------------------------------------------------
// out = agg + bias; hidden layers: ReLU then write next-layer X directly as
// f16 (the only consumer is the WMMA GEMM); final layer: f32 to d_out.
// ---------------------------------------------------------------------------
__global__ void finalize(const float* __restrict__ agg, const float* __restrict__ bias,
                         _Float16* __restrict__ xh_next, float* __restrict__ out_f32,
                         int Nn, int hidden) {
  int t = blockIdx.x * blockDim.x + threadIdx.x;
  if (t >= Nn * FDIM) return;
  float v = agg[t] + bias[t & (FDIM - 1)];
  if (hidden) {
    xh_next[t] = (_Float16)fmaxf(v, 0.0f);
  } else {
    out_f32[t] = v;
  }
}

// ---------------------------------------------------------------------------
extern "C" void kernel_launch(void* const* d_in, const int* in_sizes, int n_in,
                              void* d_out, int out_size, void* d_ws, size_t ws_size,
                              hipStream_t stream) {
  const int Nn   = in_sizes[0] / FDIM;        // 50000
  const int Etot = in_sizes[1] / 2;           // 850000

  const float* x0  = (const float*)d_in[0];
  const int*   ei  = (const int*)d_in[1];
  const int*   src = ei;
  const int*   dst = ei + Etot;

  // Workspace layout (256B-aligned slices)
  char* ws = (char*)d_ws;
  size_t off = 0;
  auto alloc = [&](size_t bytes) {
    void* p = ws + off;
    off += (bytes + 255) & ~(size_t)255;
    return p;
  };
  _Float16* xh    = (_Float16*)alloc((size_t)Nn * FDIM * sizeof(_Float16));
  _Float16* bp    = (_Float16*)alloc((size_t)8 * 32 * 4 * 16 * sizeof(_Float16));
  float*    buf_h = (float*)   alloc((size_t)Nn * FDIM * sizeof(float));
  float*    agg   = (float*)   alloc((size_t)Nn * FDIM * sizeof(float));
  float*    als   = (float*)   alloc((size_t)Nn * HEADS * sizeof(float));
  float*    ald   = (float*)   alloc((size_t)Nn * HEADS * sizeof(float));
  unsigned* m_e   = (unsigned*)alloc((size_t)Nn * HEADS * sizeof(float));
  float*    s_sum = (float*)   alloc((size_t)Nn * HEADS * sizeof(float));
  float4*   expe4 = (float4*)  alloc((size_t)Etot * HEADS * sizeof(float));

  const int TPB = 256;
  const int gNodeF = (Nn * FDIM + TPB - 1) / TPB;
  const int gNodeH = (Nn * HEADS + TPB - 1) / TPB;
  const int gEdge  = (Etot + TPB - 1) / TPB;
  const int gEdgeC = (Etot * CDIM + TPB - 1) / TPB;
  const int rowTiles = (Nn + 15) >> 4;
  const int gGemm  = (rowTiles * (FDIM / 16) + (TPB / 32) - 1) / (TPB / 32);
  const int gPackB = (8 * 32 * 4 * 16 + TPB - 1) / TPB;

  conv_x<<<gNodeF, TPB, 0, stream>>>(x0, xh, Nn * FDIM);

  for (int l = 0; l < 3; ++l) {
    const float* W    = (const float*)d_in[2 + l * 4];
    const float* Asrc = (const float*)d_in[3 + l * 4];
    const float* Adst = (const float*)d_in[4 + l * 4];
    const float* B    = (const float*)d_in[5 + l * 4];

    pack_b<<<gPackB, TPB, 0, stream>>>(W, bp);
    layer_init<<<gNodeF, TPB, 0, stream>>>(agg, m_e, s_sum, Nn);
    gemm_wmma<<<gGemm, TPB, 0, stream>>>(xh, bp, buf_h, Nn);
    attn_logits<<<gNodeH, TPB, 0, stream>>>(buf_h, Asrc, Adst, als, ald, Nn);
    edge_max<<<gEdge, TPB, 0, stream>>>(src, dst, (const float4*)als,
                                        (const float4*)ald, m_e, Etot);
    decode_max<<<gNodeH, TPB, 0, stream>>>(m_e, Nn);
    edge_expsum<<<gEdge, TPB, 0, stream>>>(src, dst, (const float4*)als,
                                           (const float4*)ald, (const float4*)m_e,
                                           s_sum, expe4, Etot);
    edge_aggregate<<<gEdgeC, TPB, 0, stream>>>(src, dst, buf_h, (const float4*)s_sum,
                                               expe4, agg, Etot);
    finalize<<<gNodeF, TPB, 0, stream>>>(agg, B, xh, (float*)d_out, Nn, l < 2 ? 1 : 0);
  }
}
